// BayesianLayer_89541478187695
// MI455X (gfx1250) — matrix operand order repaired and checked by reading, not compiled, not run on hardware
//
#include <hip/hip_runtime.h>
#include <hip/hip_bf16.h>
#include <math.h>

// ---------------------------------------------------------------------------
// BayesianLayer: out[n,o] = sum_i x[n,i] * (mu[o,i] + softplus(rho[o,i])*eps[o,i])
//                           + (bmu[o] + softplus(brho[o])*beps[o])
// N=16384, IN_F=4096, OUT_F=4096, all fp32.
//
// MI455X plan:
//  - Pass 1: materialize W in bf16 (32MB ws) + bias f32; single pass over the
//    192MB of weight inputs.
//  - Pass 2: bf16 WMMA GEMM, 128x128x32 block tile, 8 wave32 waves, 2x4 WMMA
//    accumulators per wave, double-buffered LDS.
//      * W tile: global->LDS via CDNA5 async copy (global_load_async_to_lds_b128,
//        ASYNCcnt) -- zero VGPR staging.
//      * X tile: fp32 global loads, converted to bf16 in-register (8 VGPRs),
//        stored to LDS.
//    W bf16 (32MB) stays L2-resident (192MB L2); grid ordered so each X slice
//    is reused across all 32 N-tiles through L2.
// ---------------------------------------------------------------------------

typedef __bf16 bf16_t;
typedef __attribute__((ext_vector_type(16))) __bf16 v16bf;
typedef __attribute__((ext_vector_type(8)))  __bf16 v8bf;
typedef __attribute__((ext_vector_type(4)))  __bf16 v4bf;
typedef __attribute__((ext_vector_type(8)))  float  v8f;

#define IN_F   4096
#define OUT_F  4096
#define NROWS  16384

#define TILE_M 128
#define TILE_N 128
#define TILE_K 32
#define LDT    (TILE_K + 8)   // padded LDS row (halfs): kills ds bank conflicts

__device__ __forceinline__ float softplusf(float x) {
    return fmaxf(x, 0.0f) + log1pf(expf(-fabsf(x)));   // stable log1p(exp(x))
}

// CDNA5 async global->LDS copy, 16B per lane. VDST = LDS byte offset (32-bit),
// ADDR = 64-bit global address. Tracked by ASYNCcnt.
__device__ __forceinline__ void async_copy_b128(unsigned lds_off, const void* gaddr) {
    asm volatile("global_load_async_to_lds_b128 %0, %1, off"
                 :: "v"(lds_off), "v"(gaddr)
                 : "memory");
}

__device__ __forceinline__ void wait_async0() {
    asm volatile("s_wait_asynccnt 0x0" ::: "memory");
}

// ---------------------------------------------------------------------------
// Pass 1a: W_bf16[o*IN_F + i] = bf16(mu + softplus(rho) * eps), vectorized x4
// ---------------------------------------------------------------------------
__global__ __launch_bounds__(256)
void prep_weight_kernel(const float* __restrict__ mu,
                        const float* __restrict__ rho,
                        const float* __restrict__ ep,
                        bf16_t* __restrict__ wb)
{
    size_t i = ((size_t)blockIdx.x * blockDim.x + threadIdx.x) * 4;
    float4 m = *reinterpret_cast<const float4*>(mu  + i);
    float4 r = *reinterpret_cast<const float4*>(rho + i);
    float4 e = *reinterpret_cast<const float4*>(ep  + i);
    v4bf o;
    o[0] = (bf16_t)(m.x + softplusf(r.x) * e.x);
    o[1] = (bf16_t)(m.y + softplusf(r.y) * e.y);
    o[2] = (bf16_t)(m.z + softplusf(r.z) * e.z);
    o[3] = (bf16_t)(m.w + softplusf(r.w) * e.w);
    *reinterpret_cast<v4bf*>(wb + i) = o;
}

// ---------------------------------------------------------------------------
// Pass 1b: bias_ws[o] = bmu + softplus(brho) * beps   (fp32)
// ---------------------------------------------------------------------------
__global__ __launch_bounds__(256)
void prep_bias_kernel(const float* __restrict__ bmu,
                      const float* __restrict__ brho,
                      const float* __restrict__ bep,
                      float* __restrict__ bws)
{
    int i = blockIdx.x * blockDim.x + threadIdx.x;
    bws[i] = bmu[i] + softplusf(brho[i]) * bep[i];
}

// ---------------------------------------------------------------------------
// Pass 2: GEMM out = X * W^T + b using v_wmma_f32_16x16x32_bf16
// ---------------------------------------------------------------------------
__global__ __launch_bounds__(256, 1)
void bayes_gemm_kernel(const float*  __restrict__ X,     // [NROWS, IN_F] fp32
                       const bf16_t* __restrict__ Wb,    // [OUT_F, IN_F] bf16
                       const float*  __restrict__ Bias,  // [OUT_F] fp32
                       float*        __restrict__ Out)   // [NROWS, OUT_F] fp32
{
    __shared__ bf16_t sA[2][TILE_M * LDT];   // X tile  (bf16, padded rows)
    __shared__ bf16_t sB[2][TILE_N * LDT];   // W tile  (bf16, padded rows)

    const int tid  = threadIdx.x;
    const int lane = tid & 31;        // wave32
    const int l16  = lane & 15;
    const int g    = lane >> 4;       // half-wave group 0/1
    const int wid  = tid >> 5;        // 8 waves
    const int wm   = wid & 3;         // 4 waves along M -> 32 rows each
    const int wn   = wid >> 2;        // 2 waves along N -> 64 cols each

    const int m0 = blockIdx.y * TILE_M;   // grid.y = M tiles (outer)
    const int n0 = blockIdx.x * TILE_N;   // grid.x = N tiles (inner -> X reuse in L2)

    // A staging: thread t owns (row = t/2, 16 fp32 at (t&1)*16)
    const int sr = tid >> 1;
    const int sh = (tid & 1) * 16;

    // B async copy: thread t owns two 16B chunks (8 halfs each):
    //   chunk c in {tid, tid+256}: row = c/4, col = (c&3)*8
    const int br0 = tid >> 2;              // rows 0..63
    const int bc  = (tid & 3) * 8;
    const int br1 = br0 + 64;              // rows 64..127

    v8f acc[2][4];
#pragma unroll
    for (int ms = 0; ms < 2; ++ms)
#pragma unroll
        for (int ns = 0; ns < 4; ++ns)
#pragma unroll
            for (int v = 0; v < 8; ++v) acc[ms][ns][v] = 0.0f;

    v8bf ah0, ah1;   // 16 bf16 of X staged across the compute block (8 VGPRs)

    auto issue_b_async = [&](int kb, int buf) {
        const bf16_t* w0 = Wb + (size_t)(n0 + br0) * IN_F + kb + bc;
        const bf16_t* w1 = Wb + (size_t)(n0 + br1) * IN_F + kb + bc;
        unsigned l0 = (unsigned)(uintptr_t)&sB[buf][br0 * LDT + bc];
        unsigned l1 = (unsigned)(uintptr_t)&sB[buf][br1 * LDT + bc];
        async_copy_b128(l0, w0);
        async_copy_b128(l1, w1);
    };

    auto load_a = [&](int kb) {
        const float* xp = X + (size_t)(m0 + sr) * IN_F + kb + sh;
        float4 a0 = *reinterpret_cast<const float4*>(xp + 0);
        float4 a1 = *reinterpret_cast<const float4*>(xp + 4);
        float4 a2 = *reinterpret_cast<const float4*>(xp + 8);
        float4 a3 = *reinterpret_cast<const float4*>(xp + 12);
        ah0[0] = (bf16_t)a0.x; ah0[1] = (bf16_t)a0.y;
        ah0[2] = (bf16_t)a0.z; ah0[3] = (bf16_t)a0.w;
        ah0[4] = (bf16_t)a1.x; ah0[5] = (bf16_t)a1.y;
        ah0[6] = (bf16_t)a1.z; ah0[7] = (bf16_t)a1.w;
        ah1[0] = (bf16_t)a2.x; ah1[1] = (bf16_t)a2.y;
        ah1[2] = (bf16_t)a2.z; ah1[3] = (bf16_t)a2.w;
        ah1[4] = (bf16_t)a3.x; ah1[5] = (bf16_t)a3.y;
        ah1[6] = (bf16_t)a3.z; ah1[7] = (bf16_t)a3.w;
    };

    auto store_a = [&](int buf) {
        bf16_t* da = &sA[buf][sr * LDT + sh];
        *reinterpret_cast<v8bf*>(da + 0) = ah0;
        *reinterpret_cast<v8bf*>(da + 8) = ah1;
    };

    auto compute = [&](int buf) {
        v16bf afr[2], bfr[4];
        // A fragment (16x32 bf16): lane l16 = row M, elems 0-15 -> K=g*16+0..15
#pragma unroll
        for (int ms = 0; ms < 2; ++ms) {
            const bf16_t* base = &sA[buf][(wm * 32 + ms * 16 + l16) * LDT];
            v8bf lo = *reinterpret_cast<const v8bf*>(base + g * 16);
            v8bf hi = *reinterpret_cast<const v8bf*>(base + g * 16 + 8);
            afr[ms] = __builtin_shufflevector(lo, hi, 0, 1, 2, 3, 4, 5, 6, 7,
                                              8, 9, 10, 11, 12, 13, 14, 15);
        }
        // B fragment (32x16 bf16): lane l16 = col N, elems 0-15 -> K=g*16+0..15
#pragma unroll
        for (int ns = 0; ns < 4; ++ns) {
            const bf16_t* base = &sB[buf][(wn * 64 + ns * 16 + l16) * LDT + g * 16];
            v8bf lo = *reinterpret_cast<const v8bf*>(base + 0);
            v8bf hi = *reinterpret_cast<const v8bf*>(base + 8);
            bfr[ns] = __builtin_shufflevector(lo, hi, 0, 1, 2, 3, 4, 5, 6, 7,
                                              8, 9, 10, 11, 12, 13, 14, 15);
        }
#pragma unroll
        for (int ms = 0; ms < 2; ++ms)
#pragma unroll
            for (int ns = 0; ns < 4; ++ns)
                acc[ms][ns] = __builtin_amdgcn_wmma_f32_16x16x32_bf16(
                    false, afr[ms], false, bfr[ns], (short)0, acc[ms][ns],
                    false, false);
    };

    // ---- prologue: fill buffer 0 ----
    issue_b_async(0, 0);
    load_a(0);
    store_a(0);
    wait_async0();
    __syncthreads();

    // ---- double-buffered main loop over K ----
    const int KSTEPS = IN_F / TILE_K;   // 128
    for (int kt = 0; kt < KSTEPS; ++kt) {
        const int cur = kt & 1;
        if (kt + 1 < KSTEPS) {
            issue_b_async((kt + 1) * TILE_K, cur ^ 1);   // async -> LDS (no VGPRs)
            load_a((kt + 1) * TILE_K);                   // fp32 -> bf16 in-reg
        }
        compute(cur);
        if (kt + 1 < KSTEPS) store_a(cur ^ 1);
        wait_async0();          // async B writes visible before the barrier
        __syncthreads();
    }

    // ---- epilogue: add bias, store fp32 ----
    // C/D layout: lane col = l16, VGPR v -> row = g*8 + v
#pragma unroll
    for (int ns = 0; ns < 4; ++ns) {
        const int col = n0 + wn * 64 + ns * 16 + l16;
        const float bv = Bias[col];
#pragma unroll
        for (int ms = 0; ms < 2; ++ms) {
            const int rbase = m0 + wm * 32 + ms * 16 + g * 8;
#pragma unroll
            for (int v = 0; v < 8; ++v) {
                Out[(size_t)(rbase + v) * OUT_F + col] = acc[ms][ns][v] + bv;
            }
        }
    }
}

// ---------------------------------------------------------------------------
extern "C" void kernel_launch(void* const* d_in, const int* in_sizes, int n_in,
                              void* d_out, int out_size, void* d_ws, size_t ws_size,
                              hipStream_t stream) {
    const float* x     = (const float*)d_in[0];   // [N, IN_F]
    const float* w_mu  = (const float*)d_in[1];   // [OUT_F, IN_F]
    const float* w_rho = (const float*)d_in[2];   // [OUT_F, IN_F]
    const float* b_mu  = (const float*)d_in[3];   // [OUT_F]
    const float* b_rho = (const float*)d_in[4];   // [OUT_F]
    const float* eps_w = (const float*)d_in[5];   // [OUT_F, IN_F]
    const float* eps_b = (const float*)d_in[6];   // [OUT_F]
    float* out = (float*)d_out;

    // workspace: bf16 weight (32MB) + fp32 bias (16KB)
    bf16_t* wb  = (bf16_t*)d_ws;
    float*  bws = (float*)((char*)d_ws + (size_t)OUT_F * IN_F * sizeof(bf16_t));

    // Pass 1a: weight materialization (4 elems/thread)
    {
        const size_t total = (size_t)OUT_F * IN_F / 4;   // 4,194,304
        dim3 grid((unsigned)(total / 256)), block(256);
        prep_weight_kernel<<<grid, block, 0, stream>>>(w_mu, w_rho, eps_w, wb);
    }
    // Pass 1b: bias
    {
        dim3 grid(OUT_F / 256), block(256);
        prep_bias_kernel<<<grid, block, 0, stream>>>(b_mu, b_rho, eps_b, bws);
    }
    // Pass 2: WMMA GEMM
    {
        dim3 grid(OUT_F / TILE_N, NROWS / TILE_M);   // (32, 128), n-tiles fastest
        dim3 block(256);
        bayes_gemm_kernel<<<grid, block, 0, stream>>>(x, wb, bws, out);
    }
}
